// SymbolicAttention_60421599920396
// MI455X (gfx1250) — matrix operand order repaired and checked by reading, not compiled, not run on hardware
//
#include <hip/hip_runtime.h>
#include <hip/hip_bf16.h>
#include <math.h>

// ---------------------------------------------------------------------------
// MI455X (gfx1250) bf16-WMMA implementation of ALiBi causal attention block.
// B=2, T=2048, C=1024, H=16, D=64.
// Round 3: softmax VALU diet — hoisted ALiBi terms (no int->float converts in
// the loop), causal mask only on the diagonal block (MASKED template split),
// base-2 exp via v_exp_f32, rcp for normalization. GEMM keeps async-to-LDS
// staging (global_load_async_to_lds_b128 + s_wait_asynccnt).
// ---------------------------------------------------------------------------

typedef __attribute__((ext_vector_type(16))) __bf16 v16bf;
typedef __attribute__((ext_vector_type(8)))  float  v8f;
typedef __attribute__((ext_vector_type(8)))  unsigned int v8u;
typedef unsigned short u16;
typedef unsigned int   u32;

#define B_ 2
#define T_ 2048
#define C_ 1024
#define H_ 16
#define D_ 64
#define BT_ (B_ * T_)   // 4096
#define LOG2E_ 1.44269504088896340736f

// ---- helpers --------------------------------------------------------------

__device__ __forceinline__ u16 f2bf_raw(float f) {
  u32 u = __float_as_uint(f);
  u += 0x7fffu + ((u >> 16) & 1u);   // round-to-nearest-even
  return (u16)(u >> 16);
}

// A-fragment: 16x32 bf16 (MxK), row-major source with leading dim `ld`.
// ISA layout: lane = M; per-lane K dword-pair bases {0,2,4,6,16,18,20,22}+8*half.
__device__ __forceinline__ v16bf load_a_frag(const u16* base, int ld, int lane) {
  const int row = lane & 15, half = lane >> 4;
  const u16* p = base + row * ld + 8 * half;
  v8u w;
#pragma unroll
  for (int i = 0; i < 8; ++i) {
    const int kb = (i < 4) ? (2 * i) : (2 * i + 8);
    w[i] = *(const u32*)(p + kb);
  }
  return __builtin_bit_cast(v16bf, w);
}

// B-fragment: 32x16 bf16 (KxN). Source stored col-major: (k,n) at base[n*ld+k].
// ISA layout: lane = N; per-lane K dword-pair bases {0,2,4,6,16,18,20,22}+8*half.
__device__ __forceinline__ v16bf load_b_frag(const u16* base, int ld, int lane) {
  const int col = lane & 15, half = lane >> 4;
  const u16* p = base + col * ld + 8 * half;
  v8u w;
#pragma unroll
  for (int i = 0; i < 8; ++i) w[i] = *(const u32*)(p + ((i < 4) ? (2 * i) : (2 * i + 8)));
  return __builtin_bit_cast(v16bf, w);
}

__device__ __forceinline__ v8f wmma_bf16(v16bf a, v16bf b, v8f c) {
  return __builtin_amdgcn_wmma_f32_16x16x32_bf16(false, a, false, b,
                                                 (short)0, c, false, false);
}

// gfx1250 async copy: 16 bytes global -> LDS per lane, tracked by ASYNCcnt.
// LDS byte address = low 32 bits of the flat address (ISA aperture rule).
__device__ __forceinline__ void async_copy_b128(void* lds_ptr, const void* gptr) {
  u32 lo = (u32)(uintptr_t)lds_ptr;
  unsigned long long ga = (unsigned long long)(uintptr_t)gptr;
  asm volatile("global_load_async_to_lds_b128 %0, %1, off"
               :: "v"(lo), "v"(ga) : "memory");
}
__device__ __forceinline__ void wait_async0() {
#if __has_builtin(__builtin_amdgcn_s_wait_asynccnt)
  __builtin_amdgcn_s_wait_asynccnt(0);
#else
  asm volatile("s_wait_asynccnt 0" ::: "memory");
#endif
}

// ---- kernel 1a: fp32 -> bf16 conversion (row-major copy) ------------------

__global__ __launch_bounds__(256) void cvt_bf16_kernel(
    const float* __restrict__ in, u16* __restrict__ out, int n) {
  int i = blockIdx.x * blockDim.x + threadIdx.x;
  if (i < n) out[i] = f2bf_raw(in[i]);
}

// ---- kernel 1b: fp32 [K][N] -> bf16 [N][K] (tiled transpose+convert) ------

__global__ __launch_bounds__(256) void cvt_transpose_kernel(
    const float* __restrict__ in, u16* __restrict__ out, int K, int N) {
  __shared__ u16 tile[32][33];
  const int k0 = blockIdx.y * 32, n0 = blockIdx.x * 32;
  const int tx = threadIdx.x & 31, ty = threadIdx.x >> 5;   // 32 x 8
#pragma unroll
  for (int r = ty; r < 32; r += 8)
    tile[r][tx] = f2bf_raw(in[(size_t)(k0 + r) * N + n0 + tx]);
  __syncthreads();
#pragma unroll
  for (int r = ty; r < 32; r += 8)
    out[(size_t)(n0 + r) * K + k0 + tx] = tile[tx][r];
}

// ---- kernel 2/4: tiled bf16 WMMA GEMM -------------------------------------
// Block: 256 threads = 8 waves, block tile 64(M) x 128(N), K step 32.
// A row-major [M][K]; Bt col-major [N][K] (pre-transposed). Tiles staged with
// global_load_async_to_lds_b128. Waves 2(M) x 4(N), each 32x32 (2x2 WMMA).
// QKV=true: scatter bf16 q/k [B,H,T,D] and v transposed [B,H,D,T];
// else fp32 out + bias.

template <bool QKV>
__global__ __launch_bounds__(256) void gemm_bf16_kernel(
    const u16* __restrict__ A, const u16* __restrict__ Bt,
    const float* __restrict__ bias, float* __restrict__ outF,
    u16* __restrict__ q, u16* __restrict__ k, u16* __restrict__ v,
    int K, int N) {
  __shared__ u16 Alds[64][32];    // [m][k]
  __shared__ u16 Blds[128][32];   // [n][k]

  const int tid   = threadIdx.x;
  const int lane  = tid & 31;
  const int wid   = tid >> 5;
  const int waveM = wid >> 2;     // 0..1
  const int waveN = wid & 3;      // 0..3
  const int m0 = blockIdx.y * 64;
  const int n0 = blockIdx.x * 128;

  // per-thread staging coordinates
  const int arow = tid >> 2, acol = (tid & 3) * 8;   // A: 64 x 32, 8 elem/thr
  const int brow = tid >> 1, bcol = (tid & 1) * 16;  // B: 128 x 32, 16 elem/thr

  v8f acc[2][2] = {};

  for (int kk = 0; kk < K; kk += 32) {
    async_copy_b128(&Alds[arow][acol],
                    &A[(size_t)(m0 + arow) * K + kk + acol]);
    async_copy_b128(&Blds[brow][bcol],
                    &Bt[(size_t)(n0 + brow) * K + kk + bcol]);
    async_copy_b128(&Blds[brow][bcol + 8],
                    &Bt[(size_t)(n0 + brow) * K + kk + bcol + 8]);
    wait_async0();
    __syncthreads();

    v16bf a0 = load_a_frag(&Alds[waveM * 32][0],      32, lane);
    v16bf a1 = load_a_frag(&Alds[waveM * 32 + 16][0], 32, lane);
    v16bf b0 = load_b_frag(&Blds[waveN * 32][0],      32, lane);
    v16bf b1 = load_b_frag(&Blds[waveN * 32 + 16][0], 32, lane);
    acc[0][0] = wmma_bf16(a0, b0, acc[0][0]);
    acc[0][1] = wmma_bf16(a0, b1, acc[0][1]);
    acc[1][0] = wmma_bf16(a1, b0, acc[1][0]);
    acc[1][1] = wmma_bf16(a1, b1, acc[1][1]);
    __syncthreads();
  }

  // epilogue: C layout M = r + 8*half, N = lane&15
  const int half = lane >> 4;
  const int nl   = lane & 15;
#pragma unroll
  for (int i = 0; i < 2; ++i) {
#pragma unroll
    for (int j = 0; j < 2; ++j) {
      const int n  = n0 + waveN * 32 + j * 16 + nl;
      const float bv = bias[n];
#pragma unroll
      for (int r = 0; r < 8; ++r) {
        const int m = m0 + waveM * 32 + i * 16 + r + 8 * half;
        const float val = acc[i][j][r] + bv;
        if constexpr (QKV) {
          const int b   = m >> 11, t = m & (T_ - 1);
          const int sel = n >> 10, rem = n & (C_ - 1);
          const int h   = rem >> 6, d = rem & (D_ - 1);
          if (sel == 2) {        // V stored transposed: [B,H,D,T]
            v[((size_t)(b * H_ + h) * D_ + d) * T_ + t] = f2bf_raw(val);
          } else {               // Q, K stored [B,H,T,D]
            u16* dst = (sel == 0) ? q : k;
            dst[((size_t)(b * H_ + h) * T_ + t) * D_ + d] = f2bf_raw(val);
          }
        } else {
          outF[(size_t)m * N + n] = val;
        }
      }
    }
  }
}

// ---- attention key-block body (MASKED folded at compile time) -------------
// All softmax math is in base-2: sc2 = scale*log2e, aq2[r] = slope2*q_row.

template <bool MASKED>
__device__ __forceinline__ void attn_block(
    int j0, int qbase, int lane, int half, int nl,
    const u16* __restrict__ Kh, const u16* __restrict__ Vh,
    u16 (*Plds)[32], v16bf qa0, v16bf qa1,
    float sc2, float slope2, const float* aq2,
    v8f (&o)[4], float (&mrow)[8], float (&lrow)[8]) {
  // S = Q K^T : 2 key tiles x 2 K-steps (K is [t][d] -> direct B-frag)
  v16bf kb0a = load_b_frag(Kh + (size_t)j0 * D_,              D_, lane);
  v16bf kb0b = load_b_frag(Kh + (size_t)j0 * D_ + 32,         D_, lane);
  v16bf kb1a = load_b_frag(Kh + (size_t)(j0 + 16) * D_,       D_, lane);
  v16bf kb1b = load_b_frag(Kh + (size_t)(j0 + 16) * D_ + 32,  D_, lane);
  v8f s0 = {}; s0 = wmma_bf16(qa0, kb0a, s0); s0 = wmma_bf16(qa1, kb0b, s0);
  v8f s1 = {}; s1 = wmma_bf16(qa0, kb1a, s1); s1 = wmma_bf16(qa1, kb1b, s1);

  const int   key0  = j0 + nl;
  const int   key1  = j0 + 16 + nl;
  const float skey0 = slope2 * (float)key0;   // per-block, per-lane constant
  const float skey1 = slope2 * (float)key1;

#pragma unroll
  for (int r = 0; r < 8; ++r) {
    float v0 = fmaf(s0[r], sc2, skey0 - aq2[r]);
    float v1 = fmaf(s1[r], sc2, skey1 - aq2[r]);
    if constexpr (MASKED) {
      const int qg = qbase + r;
      if (key0 > qg) v0 = -1.0e30f;
      if (key1 > qg) v1 = -1.0e30f;
    }
    float mt = fmaxf(v0, v1);
#pragma unroll
    for (int off = 8; off >= 1; off >>= 1)
      mt = fmaxf(mt, __shfl_xor(mt, off, 32));
    const float mnew  = fmaxf(mrow[r], mt);
    const float alpha = __builtin_amdgcn_exp2f(mrow[r] - mnew);
    const float p0 = __builtin_amdgcn_exp2f(v0 - mnew);
    const float p1 = __builtin_amdgcn_exp2f(v1 - mnew);
    float ssum = p0 + p1;
#pragma unroll
    for (int off = 8; off >= 1; off >>= 1)
      ssum += __shfl_xor(ssum, off, 32);
    lrow[r] = lrow[r] * alpha + ssum;
    mrow[r] = mnew;
#pragma unroll
    for (int nd = 0; nd < 4; ++nd) o[nd][r] *= alpha;
    Plds[r + 8 * half][nl]      = f2bf_raw(p0);
    Plds[r + 8 * half][nl + 16] = f2bf_raw(p1);
  }
  __syncthreads();

  // O += P V : P A-frag from LDS; V B-frags direct from global [d][t]
  v16bf pa = load_a_frag(&Plds[0][0], 32, lane);
#pragma unroll
  for (int nd = 0; nd < 4; ++nd) {
    v16bf vb = load_b_frag(Vh + (size_t)(nd * 16) * T_ + j0, T_, lane);
    o[nd] = wmma_bf16(pa, vb, o[nd]);
  }
  __syncthreads();
}

// ---- kernel 3: flash attention, one wave per (b, h, 16-query tile) --------

__global__ __launch_bounds__(32) void attn_kernel(
    const u16* __restrict__ Q, const u16* __restrict__ Kt,
    const u16* __restrict__ Vt, u16* __restrict__ Y,
    const float* __restrict__ temperature) {
  __shared__ u16 Plds[16][32];   // [q][key]

  const int lane = threadIdx.x;
  const int half = lane >> 4;
  const int nl   = lane & 15;
  const int qt   = blockIdx.x;   // query tile (16 rows)
  const int h    = blockIdx.y;
  const int b    = blockIdx.z;

  const size_t headBase = (size_t)(b * H_ + h) * T_ * D_;
  const u16* Qh = Q  + headBase;
  const u16* Kh = Kt + headBase;
  const u16* Vh = Vt + headBase;   // [d][t]

  const float sc2    = (0.125f / temperature[0]) * LOG2E_;
  const float slope2 = exp2f(-0.5f * (float)(h + 1)) * LOG2E_;

  // Q fragments: 16 rows x 64 d, split into two K=32 A-fragments
  v16bf qa0 = load_a_frag(Qh + (size_t)qt * 16 * D_,      D_, lane);
  v16bf qa1 = load_a_frag(Qh + (size_t)qt * 16 * D_ + 32, D_, lane);

  v8f   o[4] = {};
  float mrow[8], lrow[8], aq2[8];
  const int qbase = qt * 16 + 8 * half;
#pragma unroll
  for (int r = 0; r < 8; ++r) {
    mrow[r] = -1.0e30f;
    lrow[r] = 0.0f;
    aq2[r]  = slope2 * (float)(qbase + r);   // hoisted ALiBi row term
  }

  const int qlo   = qt * 16;
  const int nkb   = (qlo + 15) / 32 + 1;                       // needed blocks
  const int nfull = (qlo >= 31) ? ((qlo - 31) / 32 + 1) : 0;   // unmasked ones

  for (int kb = 0; kb < nfull; ++kb)
    attn_block<false>(kb * 32, qbase, lane, half, nl, Kh, Vh, Plds,
                      qa0, qa1, sc2, slope2, aq2, o, mrow, lrow);
  for (int kb = nfull; kb < nkb; ++kb)
    attn_block<true>(kb * 32, qbase, lane, half, nl, Kh, Vh, Plds,
                     qa0, qa1, sc2, slope2, aq2, o, mrow, lrow);

  // normalize + store y as bf16 in [B,T,C] for the projection GEMM
#pragma unroll
  for (int r = 0; r < 8; ++r) {
    const int qg = qlo + r + 8 * half;
    const float inv = __builtin_amdgcn_rcpf(lrow[r]);
    const size_t rowBase = (size_t)(b * T_ + qg) * C_ + h * D_;
#pragma unroll
    for (int nd = 0; nd < 4; ++nd)
      Y[rowBase + nd * 16 + nl] = f2bf_raw(o[nd][r] * inv);
  }
}

// ---- launcher -------------------------------------------------------------

extern "C" void kernel_launch(void* const* d_in, const int* in_sizes, int n_in,
                              void* d_out, int out_size, void* d_ws, size_t ws_size,
                              hipStream_t stream) {
  const float* x      = (const float*)d_in[0];
  const float* W_attn = (const float*)d_in[1];
  const float* b_attn = (const float*)d_in[2];
  const float* W_proj = (const float*)d_in[3];
  const float* b_proj = (const float*)d_in[4];
  const float* temp   = (const float*)d_in[5];
  float* out = (float*)d_out;

  char* p = (char*)d_ws;
  auto take = [&](size_t bytes) -> char* {
    char* r = p;
    p += (bytes + 255) & ~(size_t)255;
    return r;
  };
  u16* xb  = (u16*)take((size_t)BT_ * C_ * 2);        // x bf16 [M][K]
  u16* Wat = (u16*)take((size_t)3 * C_ * C_ * 2);     // W_attn^T bf16 [3C][C]
  u16* Wpt = (u16*)take((size_t)C_ * C_ * 2);         // W_proj^T bf16 [C][C]
  u16* Qb  = (u16*)take((size_t)B_ * H_ * T_ * D_ * 2);   // [B,H,T,D]
  u16* Kb  = (u16*)take((size_t)B_ * H_ * T_ * D_ * 2);   // [B,H,T,D]
  u16* Vb  = (u16*)take((size_t)B_ * H_ * T_ * D_ * 2);   // [B,H,D,T]
  u16* Yb  = (u16*)take((size_t)BT_ * C_ * 2);        // y bf16 [M][K]

  // 1) conversions: x row-major; W transposed to [N][K] for vector B staging
  {
    int n = BT_ * C_;
    cvt_bf16_kernel<<<n / 256, 256, 0, stream>>>(x, xb, n);
  }
  {
    dim3 grid(3 * C_ / 32, C_ / 32);
    cvt_transpose_kernel<<<grid, 256, 0, stream>>>(W_attn, Wat, C_, 3 * C_);
  }
  {
    dim3 grid(C_ / 32, C_ / 32);
    cvt_transpose_kernel<<<grid, 256, 0, stream>>>(W_proj, Wpt, C_, C_);
  }

  // 2) QKV GEMM: (4096 x 1024) @ (1024 x 3072) + bias -> q/k/v bf16
  {
    dim3 grid(3 * C_ / 128, BT_ / 64);
    gemm_bf16_kernel<true><<<grid, 256, 0, stream>>>(
        xb, Wat, b_attn, nullptr, Qb, Kb, Vb, C_, 3 * C_);
  }

  // 3) flash attention per (b, h, 16-query tile)
  {
    dim3 grid(T_ / 16, H_, B_);
    attn_kernel<<<grid, 32, 0, stream>>>(Qb, Kb, Vb, Yb, temp);
  }

  // 4) projection GEMM: (4096 x 1024) @ (1024 x 1024) + bias -> fp32 out
  {
    dim3 grid(C_ / 128, BT_ / 64);
    gemm_bf16_kernel<false><<<grid, 256, 0, stream>>>(
        Yb, Wpt, b_proj, out, nullptr, nullptr, nullptr, C_, C_);
  }
}
